// MultiDirectionSSM2DRefiner_2173253452156
// MI455X (gfx1250) — compile-verified, matrix operand order checked by reading, not run on hardware
//
#include <hip/hip_runtime.h>
#include <hip/hip_bf16.h>

// ---------------------------------------------------------------------------
// MultiDirectionSSM2DRefiner for MI455X (gfx1250, wave32, WMMA, TDM).
// B=4, C=64, H=W=192, L=H*W=36864, P=64.
//  * all matmuls: v_wmma_f32_16x16x32_bf16 (bf16 in, f32 accumulate),
//    weights pre-swizzled into the B-fragment lane layout (2 x b128 per frag)
//  * A tiles staged to LDS via the Tensor Data Mover when available
//    (tensor_load_to_lds + s_wait_tensorcnt), else coalesced uint4 copy
//  * S5 recurrence: 3-stage chunked scan (coalesced 512B-row reads):
//      scan_local (per-chunk reduction) -> scan_carry (wave32 shuffle scan of
//      chunk carries; closed-form chunk decay exp(lam*dt*CHL)) -> scan_emit
// ---------------------------------------------------------------------------

typedef unsigned short u16;
typedef unsigned int   u32;
typedef unsigned long long u64;
typedef __attribute__((ext_vector_type(16))) __bf16 v16bf;
typedef __attribute__((ext_vector_type(8)))  float  v8f;
typedef __attribute__((ext_vector_type(4)))  u32    u32x4;
typedef __attribute__((ext_vector_type(8)))  int    i32x8;
typedef __attribute__((ext_vector_type(4)))  int    i32x4;

#define BSZ 4
#define CCH 64
#define HH  192
#define WW  192
#define LL  (HH * WW)      // 36864
#define RR  (BSZ * LL)     // 147456 rows (b*L)
#define NCH 256            // scan chunks per sequence
#define CHL (LL / NCH)     // 144 steps per chunk

#if defined(__has_builtin)
#if __has_builtin(__builtin_amdgcn_tensor_load_to_lds)
#define HAVE_TDM 1
#endif
#endif

__device__ __forceinline__ u16 f2bf(float f) {
  u32 b = __float_as_uint(f);
  u32 r = (b + 0x7FFFu + ((b >> 16) & 1u)) >> 16;  // round-to-nearest-even
  return (u16)r;
}

__device__ __forceinline__ float gelu_f(float x) {
  float x3 = x * x * x;
  return 0.5f * x * (1.0f + tanhf(0.7978845608028654f * (x + 0.044715f * x3)));
}

// 16-bit A-matrix (16x32) halfword->K mapping per CDNA5 ISA 7.12.2.
__device__ __forceinline__ int kmap(int i, int lane) {
  int v = i >> 1;
  return ((v >= 4) ? 16 : 0) + ((v & 3) << 1) + (i & 1) + ((lane & 16) ? 8 : 0);
}

union FragB { u16 u[16]; v16bf v; uint4 q[2]; };
union FragC { float f[8]; v8f v; };
union G0U   { u32 w[4]; u32x4 v; };
union G1U   { int w[8]; i32x8 v; };
union G4U   { int w[4]; i32x4 v; };
union G8U   { int w[8]; i32x8 v; };

// Issue a TDM load of `bytes/2` bf16 elements (contiguous) into LDS offset.
__device__ __forceinline__ void tdm_load_1d(u32 lds_off, const void* gsrc, u32 nelem) {
#if defined(HAVE_TDM)
  u64 ga = (u64)(__SIZE_TYPE__)gsrc;
  G0U g0; G1U g1; G4U z4; G8U z8;
  for (int i = 0; i < 8; ++i) g1.w[i] = 0;
  for (int i = 0; i < 4; ++i) z4.w[i] = 0;
  for (int i = 0; i < 8; ++i) z8.w[i] = 0;
  g0.w[0] = 1u;                                   // count=1, user D#
  g0.w[1] = lds_off;                              // lds_addr (bytes)
  g0.w[2] = (u32)(ga & 0xffffffffu);              // global_addr[31:0]
  g0.w[3] = (u32)((ga >> 32) & 0x01ffffffu) | (2u << 30);  // addr[56:32] | type=2
  // group1: mask=0, data_size=1 (2B), no barrier/iterate/pad
  g1.w[0] = 0x00010000;
  g1.w[1] = (int)((nelem & 0xffffu) << 16);       // tensor_dim0[15:0] @bits63:48
  g1.w[2] = (int)(((nelem >> 16) & 0xffffu) | (1u << 16));  // dim0 hi | tensor_dim1=1
  g1.w[3] = (int)((nelem & 0xffffu) << 16);       // tile_dim0 @bits127:112
  g1.w[4] = 0;                                    // tile_dim1=0 (1D), tile_dim2=0
  g1.w[5] = (int)nelem;                           // tensor_dim0_stride lo
  g1.w[6] = 0; g1.w[7] = 0;
#if defined(__clang_major__) && (__clang_major__ >= 23)
  __builtin_amdgcn_tensor_load_to_lds(g0.v, g1.v, z4.v, z4.v, z8.v, 0);
#else
  __builtin_amdgcn_tensor_load_to_lds(g0.v, g1.v, z4.v, z4.v, 0);
#endif
#else
  (void)lds_off; (void)gsrc; (void)nelem;
#endif
}

// Decode linear swizzled-weight index -> (n, k) (B-fragment layout).
__device__ __forceinline__ void swz_decode(int e, int NT, int& n, int& k) {
  int blk = e >> 9, w = e & 511, lane = w >> 4, i = w & 15;
  int kc = blk / NT, nt = blk - kc * NT;
  n = nt * 16 + (lane & 15);
  k = kc * 32 + kmap(i, lane);
}

// ZOH coefficient (lam_bar - 1)/lam for B-bar.
__device__ __forceinline__ void bbar_coef(const float* lam_re, const float* lam_im,
                                          const float* log_step, int dir, int p,
                                          float& cr, float& ci) {
  float lr = lam_re[dir * 64 + p], li = lam_im[dir * 64 + p];
  float dt = expf(log_step[dir * 64 + p]);
  float er = expf(lr * dt);
  float lbr = er * cosf(li * dt), lbi = er * sinf(li * dt);
  float den = lr * lr + li * li;
  cr = ((lbr - 1.f) * lr + lbi * li) / den;
  ci = (lbi * lr - (lbr - 1.f) * li) / den;
}

// ---------------------------------------------------------------------------
// Stage 0: (b,c,h,w) -> seqH (b, h*W+w, c) and seqV (b, w*H+h, c)
// ---------------------------------------------------------------------------
__global__ __launch_bounds__(256) void prep_seq(const float* __restrict__ x,
                                                float* __restrict__ sqh,
                                                float* __restrict__ sqv) {
  size_t idx = (size_t)blockIdx.x * 256 + threadIdx.x;  // grid covers B*C*H*W
  int w = (int)(idx % WW);
  size_t t = idx / WW;
  int h = (int)(t % HH); t /= HH;
  int c = (int)(t % CCH);
  int b = (int)(t / CCH);
  float v = x[idx];
  sqh[((size_t)b * LL + h * WW + w) * 64 + c] = v;
  sqv[((size_t)b * LL + w * HH + h) * 64 + c] = v;
}

// ---------------------------------------------------------------------------
// Stage 0b: scan decays + all WMMA-swizzled bf16 weights.
// ---------------------------------------------------------------------------
__global__ void prep_weights(const float* __restrict__ lam_re, const float* __restrict__ lam_im,
                             const float* __restrict__ B_re,  const float* __restrict__ B_im,
                             const float* __restrict__ C_re,  const float* __restrict__ C_im,
                             const float* __restrict__ log_step,
                             const float* __restrict__ ff_enc, const float* __restrict__ ff_dec,
                             const float* __restrict__ fusion_w,
                             float* __restrict__ pbuf,
                             u16* __restrict__ WbSw, u16* __restrict__ WcSw,
                             u16* __restrict__ WeSw, u16* __restrict__ WdSw,
                             u16* __restrict__ WfSw) {
  const int S0 = 128;         // pbuf entries
  const int S1 = 2 * 8192;    // Wb  (K=64,N=128)
  const int S2 = 2 * 16384;   // Wc  (K=256,N=64)
  const int S3 = 2 * 4096;    // We
  const int S4 = 2 * 4096;    // Wd
  const int S5 = 8192;        // Wf  (K=128,N=64)
  const int total = S0 + S1 + S2 + S3 + S4 + S5;
  for (int e0 = blockIdx.x * blockDim.x + threadIdx.x; e0 < total;
       e0 += gridDim.x * blockDim.x) {
    int e = e0;
    if (e < S0) {
      int dir = e >> 6, p = e & 63;
      float lr = lam_re[dir * 64 + p], li = lam_im[dir * 64 + p];
      float dt = expf(log_step[dir * 64 + p]);
      float ldr = lr * dt, ldi = li * dt;
      float er = expf(ldr);
      pbuf[e * 4 + 0] = er * cosf(ldi);   // lam_bar re
      pbuf[e * 4 + 1] = er * sinf(ldi);   // lam_bar im
      pbuf[e * 4 + 2] = ldr;              // lam*dt re
      pbuf[e * 4 + 3] = ldi;              // lam*dt im
    } else if ((e -= S0) < S1) {
      int dir = e / 8192, ee = e - dir * 8192;
      int n, k; swz_decode(ee, 8, n, k);
      int p = n & 63;
      float cr, ci; bbar_coef(lam_re, lam_im, log_step, dir, p, cr, ci);
      float br = B_re[(dir * 64 + p) * 64 + k];
      float bi = B_im[(dir * 64 + p) * 64 + k];
      float val = (n < 64) ? (cr * br - ci * bi) : (cr * bi + ci * br);
      WbSw[dir * 8192 + ee] = f2bf(val);
    } else if ((e -= S1) < S2) {
      int dir = e / 16384, ee = e - dir * 16384;
      int n, k; swz_decode(ee, 4, n, k);
      float val;
      const float* cre = C_re + (size_t)(dir * 64 + n) * 128;
      const float* cim = C_im + (size_t)(dir * 64 + n) * 128;
      if      (k <  64) val =  2.f * cre[k];
      else if (k < 128) val = -2.f * cim[k - 64];
      else if (k < 192) val =  2.f * cre[64 + (k - 128)];
      else              val = -2.f * cim[64 + (k - 192)];
      WcSw[dir * 16384 + ee] = f2bf(val);
    } else if ((e -= S2) < S3) {
      int dir = e / 4096, ee = e - dir * 4096;
      int n, k; swz_decode(ee, 4, n, k);
      WeSw[dir * 4096 + ee] = f2bf(ff_enc[(size_t)(dir * 64 + n) * 64 + k]);
    } else if ((e -= S3) < S4) {
      int dir = e / 4096, ee = e - dir * 4096;
      int n, k; swz_decode(ee, 4, n, k);
      WdSw[dir * 4096 + ee] = f2bf(ff_dec[(size_t)(dir * 64 + n) * 64 + k]);
    } else {
      e -= S4;
      int n, k; swz_decode(e, 4, n, k);
      WfSw[e] = f2bf(fusion_w[(size_t)n * 128 + k]);
    }
  }
}

// ---------------------------------------------------------------------------
// LayerNorm over 64 channels: one wave per position (wave32, 2 ch per lane).
// ---------------------------------------------------------------------------
__global__ __launch_bounds__(256) void ln_kernel(const float* __restrict__ in,
                                                 const float* __restrict__ g,
                                                 const float* __restrict__ bb,
                                                 float* __restrict__ out32,
                                                 u16* __restrict__ outbf, int R) {
  int lane = threadIdx.x & 31, wave = threadIdx.x >> 5;
  int pos = blockIdx.x * 8 + wave;
  if (pos >= R) return;
  const float* row = in + (size_t)pos * 64;
  float v0 = row[lane], v1 = row[lane + 32];
  float s = v0 + v1;
  for (int d = 16; d >= 1; d >>= 1) s += __shfl_xor(s, d, 32);
  float mu = s * (1.0f / 64.0f);
  float d0 = v0 - mu, d1 = v1 - mu;
  float q = d0 * d0 + d1 * d1;
  for (int d = 16; d >= 1; d >>= 1) q += __shfl_xor(q, d, 32);
  float rstd = rsqrtf(q * (1.0f / 64.0f) + 1e-5f);
  float o0 = d0 * rstd * g[lane] + bb[lane];
  float o1 = d1 * rstd * g[lane + 32] + bb[lane + 32];
  size_t base = (size_t)pos * 64;
  out32[base + lane] = o0;      out32[base + lane + 32] = o1;
  outbf[base + lane] = f2bf(o0); outbf[base + lane + 32] = f2bf(o1);
}

// ---------------------------------------------------------------------------
// Scan stage 1: per-chunk local reduction.  One wave per (b, chunk);
// lanes hold states p=lane and p=lane+32 -> every step reads one contiguous
// 512B Bu row (fully coalesced).  Handles fw (rev=0) and bw (rev=1).
// ---------------------------------------------------------------------------
__global__ __launch_bounds__(256) void scan_local(const float* __restrict__ Bu,
                                                  const float* __restrict__ pb,
                                                  float* __restrict__ carr) {
  int lane = threadIdx.x & 31, wave = threadIdx.x >> 5;
  int wid = blockIdx.x * 8 + wave;                 // 0 .. BSZ*NCH-1
  int bidx = wid / NCH, ch = wid - bidx * NCH;
  int p0 = lane, p1 = lane + 32;
  float l0r = pb[p0 * 4 + 0], l0i = pb[p0 * 4 + 1];
  float l1r = pb[p1 * 4 + 0], l1i = pb[p1 * 4 + 1];
  const float* bu = Bu + (size_t)bidx * LL * 128;
  for (int rev = 0; rev < 2; ++rev) {
    float x0r = 0.f, x0i = 0.f, x1r = 0.f, x1i = 0.f;
    for (int j = 0; j < CHL; ++j) {
      int t = ch * CHL + j;
      int l = rev ? (LL - 1 - t) : t;
      const float* row = bu + (size_t)l * 128;
      int lp = rev ? (l >= 8 ? l - 8 : l) : (l + 8 < LL ? l + 8 : l);
      __builtin_prefetch(bu + (size_t)lp * 128 + lane, 0, 0);
      float b0r = row[p0], b0i = row[64 + p0];
      float b1r = row[p1], b1i = row[64 + p1];
      float n0r = l0r * x0r - l0i * x0i + b0r;
      float n0i = l0r * x0i + l0i * x0r + b0i;
      float n1r = l1r * x1r - l1i * x1i + b1r;
      float n1i = l1r * x1i + l1i * x1r + b1i;
      x0r = n0r; x0i = n0i; x1r = n1r; x1i = n1i;
    }
    float* cb = carr + ((size_t)(bidx * 2 + rev) * NCH + ch) * 128;
    cb[p0] = x0r; cb[64 + p0] = x0i;
    cb[p1] = x1r; cb[64 + p1] = x1i;
  }
}

// ---------------------------------------------------------------------------
// Scan stage 2: exclusive scan of chunk carries.  One wave per (b, p);
// 8 chunks per lane serially, then wave32 shuffle scan; carries rewritten
// in place as exclusive values.  Chunk decay A1 = exp(lam*dt*CHL).
// ---------------------------------------------------------------------------
__global__ __launch_bounds__(256) void scan_carry(float* __restrict__ carr,
                                                  const float* __restrict__ pb) {
  int lane = threadIdx.x & 31, wave = threadIdx.x >> 5;
  int wid = blockIdx.x * 8 + wave;                 // 0..BSZ*64-1
  int bidx = wid >> 6, p = wid & 63;
  float ldr = pb[p * 4 + 2], ldi = pb[p * 4 + 3];
  float ea  = expf(ldr * (float)CHL);
  float A1r = ea * cosf(ldi * (float)CHL);
  float A1i = ea * sinf(ldi * (float)CHL);
  for (int rev = 0; rev < 2; ++rev) {
    float* cb = carr + (size_t)(bidx * 2 + rev) * NCH * 128;
    // lane-local sequential combine over its 8 chunks
    float Ar = 1.f, Ai = 0.f, br = 0.f, bi = 0.f;
    for (int i = 0; i < NCH / 32; ++i) {
      int ch = lane * (NCH / 32) + i;
      float cr = cb[(size_t)ch * 128 + p], ci = cb[(size_t)ch * 128 + 64 + p];
      float nAr = Ar * A1r - Ai * A1i;
      float nAi = Ar * A1i + Ai * A1r;
      float nbr = A1r * br - A1i * bi + cr;
      float nbi = A1r * bi + A1i * br + ci;
      Ar = nAr; Ai = nAi; br = nbr; bi = nbi;
    }
    // wave32 inclusive shuffle scan:  binop(prev,cur) = (Ap*Ac, Ac*bp + bc)
    for (int d = 1; d < 32; d <<= 1) {
      float pAr = __shfl_up(Ar, d), pAi = __shfl_up(Ai, d);
      float pbr = __shfl_up(br, d), pbi = __shfl_up(bi, d);
      if (lane >= d) {
        float nAr = pAr * Ar - pAi * Ai;
        float nAi = pAr * Ai + pAi * Ar;
        float nbr = Ar * pbr - Ai * pbi + br;
        float nbi = Ar * pbi + Ai * pbr + bi;
        Ar = nAr; Ai = nAi; br = nbr; bi = nbi;
      }
    }
    float exr = __shfl_up(br, 1), exi = __shfl_up(bi, 1);
    if (lane == 0) { exr = 0.f; exi = 0.f; }
    // rewalk, writing the exclusive carry entering each chunk (in place)
    float xr = exr, xi = exi;
    for (int i = 0; i < NCH / 32; ++i) {
      int ch = lane * (NCH / 32) + i;
      float cr = cb[(size_t)ch * 128 + p], ci = cb[(size_t)ch * 128 + 64 + p];
      cb[(size_t)ch * 128 + p] = xr;
      cb[(size_t)ch * 128 + 64 + p] = xi;
      float nxr = A1r * xr - A1i * xi + cr;
      float nxi = A1r * xi + A1i * xr + ci;
      xr = nxr; xi = nxi;
    }
  }
}

// ---------------------------------------------------------------------------
// Scan stage 3: replay with exclusive carries, emitting bf16 state rows
// xs[l] = [xf_re(64) | xf_im(64) | xb_re(64) | xb_im(64)]  (coalesced).
// ---------------------------------------------------------------------------
__global__ __launch_bounds__(256) void scan_emit(const float* __restrict__ Bu,
                                                 const float* __restrict__ pb,
                                                 const float* __restrict__ carr,
                                                 u16* __restrict__ xs) {
  int lane = threadIdx.x & 31, wave = threadIdx.x >> 5;
  int wid = blockIdx.x * 8 + wave;
  int bidx = wid / NCH, ch = wid - bidx * NCH;
  int p0 = lane, p1 = lane + 32;
  float l0r = pb[p0 * 4 + 0], l0i = pb[p0 * 4 + 1];
  float l1r = pb[p1 * 4 + 0], l1i = pb[p1 * 4 + 1];
  const float* bu = Bu + (size_t)bidx * LL * 128;
  u16* xb = xs + (size_t)bidx * LL * 256;
  for (int rev = 0; rev < 2; ++rev) {
    const float* cb = carr + ((size_t)(bidx * 2 + rev) * NCH + ch) * 128;
    float x0r = cb[p0], x0i = cb[64 + p0];
    float x1r = cb[p1], x1i = cb[64 + p1];
    const int o = rev ? 128 : 0;
    for (int j = 0; j < CHL; ++j) {
      int t = ch * CHL + j;
      int l = rev ? (LL - 1 - t) : t;
      const float* row = bu + (size_t)l * 128;
      int lp = rev ? (l >= 8 ? l - 8 : l) : (l + 8 < LL ? l + 8 : l);
      __builtin_prefetch(bu + (size_t)lp * 128 + lane, 0, 0);
      float b0r = row[p0], b0i = row[64 + p0];
      float b1r = row[p1], b1i = row[64 + p1];
      float n0r = l0r * x0r - l0i * x0i + b0r;
      float n0i = l0r * x0i + l0i * x0r + b0i;
      float n1r = l1r * x1r - l1i * x1i + b1r;
      float n1i = l1r * x1i + l1i * x1r + b1i;
      x0r = n0r; x0i = n0i; x1r = n1r; x1i = n1i;
      u16* xrow = xb + (size_t)l * 256 + o;
      xrow[p0] = f2bf(x0r); xrow[64 + p0] = f2bf(x0i);
      xrow[p1] = f2bf(x1r); xrow[64 + p1] = f2bf(x1i);
    }
  }
}

// ---------------------------------------------------------------------------
// Generic WMMA GEMM: out[m,n] = sum_k A[m,k]*W[n,k]; 16-row tile per wave,
// 8 waves/block.  A tile -> LDS via TDM (or uint4 copy fallback); W is
// pre-swizzled into the B-fragment layout.
// MODE 0: fp32 store     MODE 1: h=acc+D[n]*fx; out=gelu(h)+fx (fp32)
// MODE 2: bf16 store     MODE 3: out = acc + aux0 residual (bf16)
// ---------------------------------------------------------------------------
template <int K, int N, int MODE>
__global__ __launch_bounds__(256) void gemm_tile(const u16* __restrict__ Abf,
                                                 const u16* __restrict__ Wsw,
                                                 void* __restrict__ outp,
                                                 const float* __restrict__ aux0,
                                                 const float* __restrict__ aux1,
                                                 int R) {
  extern __shared__ u16 smem[];
  const int lane = threadIdx.x & 31;
  const int wave = threadIdx.x >> 5;
  const int tile = blockIdx.x * 8 + wave;
  u16* at = smem + wave * (16 * K);
  const int r0 = tile * 16;
  const bool active = (r0 < R);
#if defined(HAVE_TDM)
  if (active)
    tdm_load_1d((u32)(wave * 16 * K * 2), Abf + (size_t)r0 * K, (u32)(16 * K));
  __builtin_amdgcn_s_wait_tensorcnt(0);
#else
  if (active) {  // contiguous 16xK bf16 tile copy
    const uint4* src = (const uint4*)(Abf + (size_t)r0 * K);
    uint4* dst = (uint4*)at;
    for (int i = lane; i < (16 * K) / 8; i += 32) dst[i] = src[i];
  }
#endif
  __syncthreads();
  if (!active) return;

  constexpr int KT = K / 32, NT = N / 16;
  FragC acc[NT];
  for (int nt = 0; nt < NT; ++nt)
    for (int r = 0; r < 8; ++r) acc[nt].f[r] = 0.f;

  for (int kc = 0; kc < KT; ++kc) {
    FragB a;
    const int m = lane & 15;
    for (int i = 0; i < 16; ++i) a.u[i] = at[m * K + kc * 32 + kmap(i, lane)];
    for (int nt = 0; nt < NT; ++nt) {
      FragB b;
      const uint4* wsrc = (const uint4*)(Wsw + ((size_t)(kc * NT + nt) * 32 + lane) * 16);
      b.q[0] = wsrc[0]; b.q[1] = wsrc[1];
      acc[nt].v = __builtin_amdgcn_wmma_f32_16x16x32_bf16(
          false, a.v, false, b.v, (short)0, acc[nt].v, false, false);
    }
  }

  const int mb = (lane >= 16) ? 8 : 0;
  const int nc = lane & 15;
  for (int nt = 0; nt < NT; ++nt) {
    int n = nt * 16 + nc;
    for (int r = 0; r < 8; ++r) {
      size_t row = (size_t)(r0 + mb + r);
      float val = acc[nt].f[r];
      if constexpr (MODE == 0) {
        ((float*)outp)[row * N + n] = val;
      } else if constexpr (MODE == 1) {
        float fx = aux0[row * 64 + n];
        float h = val + aux1[n] * fx;           // + D*u
        ((float*)outp)[row * 64 + n] = gelu_f(h) + fx;
      } else if constexpr (MODE == 2) {
        ((u16*)outp)[row * N + n] = f2bf(val);
      } else if constexpr (MODE == 3) {
        ((u16*)outp)[row * N + n] = f2bf(val + aux0[row * 64 + n]);
      }
    }
  }
}

// ---------------------------------------------------------------------------
// Fusion: gather [yh | yv(w*H+h)] (K=128) into LDS, WMMA with fusion_w,
// write fp32 output transposed to (b,c,h,w) + bias.
// ---------------------------------------------------------------------------
__global__ __launch_bounds__(256) void fusion_kernel(const u16* __restrict__ yh,
                                                     const u16* __restrict__ yv,
                                                     const u16* __restrict__ Wf,
                                                     const float* __restrict__ bias,
                                                     float* __restrict__ out) {
  extern __shared__ u16 smemf[];
  const int lane = threadIdx.x & 31;
  const int wave = threadIdx.x >> 5;
  const int tile = blockIdx.x * 8 + wave;
  u16* at = smemf + wave * 16 * 128;
  const int r0 = tile * 16;
  const int bidx = r0 / LL;
  const int rem0 = r0 - bidx * LL;
  for (int e = lane; e < 16 * 64; e += 32) {  // 16 rows x 64 dword copies
    int m = e >> 6, c2 = e & 63;
    int rem = rem0 + m;
    int h = rem / WW, w = rem - h * WW;
    const u16* srcrow; int cc;
    if (c2 < 32) { srcrow = yh + ((size_t)bidx * LL + rem) * 64;          cc = c2 * 2; }
    else         { srcrow = yv + ((size_t)bidx * LL + (w * HH + h)) * 64; cc = (c2 - 32) * 2; }
    ((u32*)(at + m * 128))[c2] = *(const u32*)(srcrow + cc);
  }
  __syncthreads();

  FragC acc[4];
  for (int nt = 0; nt < 4; ++nt)
    for (int r = 0; r < 8; ++r) acc[nt].f[r] = 0.f;
  for (int kc = 0; kc < 4; ++kc) {
    FragB a;
    const int m = lane & 15;
    for (int i = 0; i < 16; ++i) a.u[i] = at[m * 128 + kc * 32 + kmap(i, lane)];
    for (int nt = 0; nt < 4; ++nt) {
      FragB b;
      const uint4* wsrc = (const uint4*)(Wf + ((size_t)(kc * 4 + nt) * 32 + lane) * 16);
      b.q[0] = wsrc[0]; b.q[1] = wsrc[1];
      acc[nt].v = __builtin_amdgcn_wmma_f32_16x16x32_bf16(
          false, a.v, false, b.v, (short)0, acc[nt].v, false, false);
    }
  }
  const int mb = (lane >= 16) ? 8 : 0;
  const int nc = lane & 15;
  for (int nt = 0; nt < 4; ++nt) {
    int n = nt * 16 + nc;
    float bv = bias[n];
    for (int r = 0; r < 8; ++r) {
      int rem = rem0 + mb + r;
      int h = rem / WW, w = rem - h * WW;
      out[(((size_t)bidx * CCH + n) * HH + h) * WW + w] = acc[nt].f[r] + bv;
    }
  }
}

// ---------------------------------------------------------------------------
extern "C" void kernel_launch(void* const* d_in, const int* in_sizes, int n_in,
                              void* d_out, int out_size, void* d_ws, size_t ws_size,
                              hipStream_t stream) {
  const float* x        = (const float*)d_in[0];
  const float* ln1_g    = (const float*)d_in[1];
  const float* ln1_b    = (const float*)d_in[2];
  const float* lam_re   = (const float*)d_in[3];
  const float* lam_im   = (const float*)d_in[4];
  const float* B_re     = (const float*)d_in[5];
  const float* B_im     = (const float*)d_in[6];
  const float* C_re     = (const float*)d_in[7];
  const float* C_im     = (const float*)d_in[8];
  const float* Dv       = (const float*)d_in[9];
  const float* log_step = (const float*)d_in[10];
  const float* ln2_g    = (const float*)d_in[11];
  const float* ln2_b    = (const float*)d_in[12];
  const float* ff_enc   = (const float*)d_in[13];
  const float* ff_dec   = (const float*)d_in[14];
  const float* fusion_w = (const float*)d_in[15];
  const float* fusion_b = (const float*)d_in[16];
  float* out = (float*)d_out;
  (void)in_sizes; (void)n_in; (void)out_size; (void)ws_size;

  char* base = (char*)d_ws;
  size_t off = 0;
  auto carve = [&](size_t bytes) -> char* {
    char* p = base + off;
    off += (bytes + 255) & ~(size_t)255;
    return p;
  };
  const size_t NA = (size_t)RR * CCH;                  // 9,437,184 elements
  float* seqH = (float*)carve(NA * 4);
  float* seqV = (float*)carve(NA * 4);
  float* fx32 = (float*)carve(NA * 4);                 // fx, later fx2 (fp32)
  u16*   bfA  = (u16*)  carve(NA * 2);                 // fx bf16, later t bf16
  u16*   bfB  = (u16*)  carve(NA * 2);                 // fx2 bf16
  float* Bu   = (float*)carve((size_t)RR * 128 * 4);   // Bu; later x1 fp32
  u16*   xs   = (u16*)  carve((size_t)RR * 256 * 2);   // scan states bf16
  u16*   yh   = (u16*)  carve(NA * 2);
  u16*   yv   = (u16*)  carve(NA * 2);
  float* carr = (float*)carve((size_t)BSZ * 2 * NCH * 128 * 4);  // chunk carries
  float* pbuf = (float*)carve(2 * 64 * 4 * 4);
  u16* WbSw   = (u16*)carve(2 * 8192 * 2);
  u16* WcSw   = (u16*)carve(2 * 16384 * 2);
  u16* WeSw   = (u16*)carve(2 * 4096 * 2);
  u16* WdSw   = (u16*)carve(2 * 4096 * 2);
  u16* WfSw   = (u16*)carve(8192 * 2);

  prep_seq<<<(int)(NA / 256), 256, 0, stream>>>(x, seqH, seqV);
  prep_weights<<<289, 256, 0, stream>>>(lam_re, lam_im, B_re, B_im, C_re, C_im,
                                        log_step, ff_enc, ff_dec, fusion_w,
                                        pbuf, WbSw, WcSw, WeSw, WdSw, WfSw);

  float* x1 = Bu;  // alias: Bu dead once the scan consumed it
  for (int dir = 0; dir < 2; ++dir) {
    const float* seq = dir ? seqV : seqH;
    const float* pb = pbuf + dir * 256;
    // LN1 -> fx (fp32 + bf16)
    ln_kernel<<<RR / 8, 256, 0, stream>>>(seq, ln1_g + dir * 64, ln1_b + dir * 64,
                                          fx32, bfA, RR);
    // Bu = fx @ B_bar^T  (K=64, N=128 [re|im])
    gemm_tile<64, 128, 0><<<RR / 128, 256, 8 * 16 * 64 * 2, stream>>>(
        bfA, WbSw + dir * 8192, (void*)Bu, nullptr, nullptr, RR);
    // 3-stage bidirectional chunked scan -> xs bf16
    scan_local<<<BSZ * NCH / 8, 256, 0, stream>>>(Bu, pb, carr);
    scan_carry<<<BSZ * 64 / 8, 256, 0, stream>>>(carr, pb);
    scan_emit<<<BSZ * NCH / 8, 256, 0, stream>>>(Bu, pb, carr, xs);
    // x1 = gelu(2*Re(xs@Cc^T) + D*fx) + fx   (K=256, N=64)
    gemm_tile<256, 64, 1><<<RR / 128, 256, 8 * 16 * 256 * 2, stream>>>(
        xs, WcSw + dir * 16384, (void*)x1, fx32, Dv + dir * 64, RR);
    // LN2 -> fx2 (fp32 + bf16)
    ln_kernel<<<RR / 8, 256, 0, stream>>>(x1, ln2_g + dir * 64, ln2_b + dir * 64,
                                          fx32, bfB, RR);
    // t = fx2 @ ff_enc^T
    gemm_tile<64, 64, 2><<<RR / 128, 256, 8 * 16 * 64 * 2, stream>>>(
        bfB, WeSw + dir * 4096, (void*)bfA, nullptr, nullptr, RR);
    // y_dir = t @ ff_dec^T + fx2
    gemm_tile<64, 64, 3><<<RR / 128, 256, 8 * 16 * 64 * 2, stream>>>(
        bfA, WdSw + dir * 4096, (void*)(dir ? yv : yh), fx32, nullptr, RR);
  }
  // out = fusion_w @ [y_h ; y_v] + bias, transposed to (b,c,h,w)
  fusion_kernel<<<RR / 128, 256, 8 * 16 * 128 * 2, stream>>>(yh, yv, WfSw,
                                                             fusion_b, out);
}
